// Block_48627619725752
// MI455X (gfx1250) — matrix operand order repaired and checked by reading, not compiled, not run on hardware
//
#include <hip/hip_runtime.h>
#include <hip/hip_bf16.h>
#include <cstdint>

typedef _Float16 half_t;
typedef _Float16 v16h __attribute__((ext_vector_type(16)));
typedef float    v8f  __attribute__((ext_vector_type(8)));
typedef unsigned int uint32;

// Problem constants (match reference)
#define BB_   2
#define TT_   2048
#define CC_   1024
#define HH_   16
#define DD_   64
#define MM_   (BB_ * TT_)   // 4096 token rows

// ---------------------------------------------------------------------------
// CDNA5 async copy helpers (ASYNCcnt-tracked global->LDS DMA, ISA ch.10/15.18)
// ---------------------------------------------------------------------------
__device__ __forceinline__ void async_copy_b128(const half_t* gptr, half_t* lptr) {
  const unsigned lds = (unsigned)(uintptr_t)lptr;                 // DS-domain addr
  const unsigned long long ga = (unsigned long long)(uintptr_t)gptr;
  asm volatile("global_load_async_to_lds_b128 %0, %1, off"
               :: "v"(lds), "v"(ga) : "memory");
}
__device__ __forceinline__ void async_wait0() {
  asm volatile("s_wait_asynccnt 0x0" ::: "memory");
}

// ---------------------------------------------------------------------------
// WMMA fragment loaders from LDS (contiguous k = 16*(lane/16) + s in both A and B, wave32)
// ---------------------------------------------------------------------------
__device__ __forceinline__ v16h frag_a_lds(const half_t* base, int stride) {
  const int lane = threadIdx.x & 31;
  const int r = lane & 15, hsel = lane >> 4;
  const half_t* rp = base + r * stride;
  union { v16h v; uint32 u[8]; } f;
#pragma unroll
  for (int j = 0; j < 8; ++j) {
    const int kb = (hsel << 4) | (j << 1);
    f.u[j] = *(const uint32*)(rp + kb);
  }
  return f.v;
}

// B fragment, LDS holds the matrix as [n][k] (N-major):
__device__ __forceinline__ v16h frag_b_lds(const half_t* base, int stride) {
  const int lane = threadIdx.x & 31;
  const int r = lane & 15, hsel = lane >> 4;
  const half_t* rp = base + r * stride;
  union { v16h v; uint32 u[8]; } f;
#pragma unroll
  for (int j = 0; j < 8; ++j) {
    const int kb = (hsel << 4) | (j << 1);
    f.u[j] = *(const uint32*)(rp + kb);
  }
  return f.v;
}

// B fragment when LDS holds the matrix K-major ([k][n], e.g. V tiles):
__device__ __forceinline__ v16h frag_b_strided_lds(const half_t* base, int stride) {
  const int lane = threadIdx.x & 31;
  const int r = lane & 15, hsel = lane >> 4;
  v16h v;
#pragma unroll
  for (int j = 0; j < 8; ++j) {
    const int k0 = (hsel << 4) | (j << 1);
    v[2 * j]     = base[(k0 + 0) * stride + r];
    v[2 * j + 1] = base[(k0 + 1) * stride + r];
  }
  return v;
}

// ---------------------------------------------------------------------------
// Fused transpose + f32->f16 cast: W[K,N] f32 -> Wt[N,K] f16 (LDS 32x32 tiles)
// ---------------------------------------------------------------------------
__global__ __launch_bounds__(256) void transpose_to_f16(const float* __restrict__ src,
                                                        half_t* __restrict__ dst,
                                                        int K, int N) {
  __shared__ float tile[32][33];
  const int k0 = blockIdx.y * 32, n0 = blockIdx.x * 32;
  const int tx = threadIdx.x, ty = threadIdx.y;   // (32,8)
#pragma unroll
  for (int i = ty; i < 32; i += 8)
    tile[i][tx] = src[(size_t)(k0 + i) * N + n0 + tx];
  __syncthreads();
#pragma unroll
  for (int i = ty; i < 32; i += 8)
    dst[(size_t)(n0 + i) * K + k0 + tx] = (half_t)tile[tx][i];
}

// ---------------------------------------------------------------------------
// LayerNorm over C=1024, one block per row, output f16
// ---------------------------------------------------------------------------
__global__ __launch_bounds__(256) void ln_kernel(const float* __restrict__ x,
                                                 const float* __restrict__ g,
                                                 const float* __restrict__ be,
                                                 half_t* __restrict__ out) {
  __shared__ float red[8];
  const int row = blockIdx.x;
  const int tid = threadIdx.x;
  const float* xr = x + (size_t)row * CC_;
  const float4 v = *(const float4*)(xr + tid * 4);
  float s = v.x + v.y + v.z + v.w;
#pragma unroll
  for (int off = 16; off >= 1; off >>= 1) s += __shfl_xor(s, off, 32);
  if ((tid & 31) == 0) red[tid >> 5] = s;
  __syncthreads();
  float tot = 0.f;
#pragma unroll
  for (int i = 0; i < 8; ++i) tot += red[i];
  const float mu = tot * (1.0f / CC_);
  const float d0 = v.x - mu, d1 = v.y - mu, d2 = v.z - mu, d3 = v.w - mu;
  float s2 = d0 * d0 + d1 * d1 + d2 * d2 + d3 * d3;
#pragma unroll
  for (int off = 16; off >= 1; off >>= 1) s2 += __shfl_xor(s2, off, 32);
  __syncthreads();
  if ((tid & 31) == 0) red[tid >> 5] = s2;
  __syncthreads();
  float tot2 = 0.f;
#pragma unroll
  for (int i = 0; i < 8; ++i) tot2 += red[i];
  const float rs = rsqrtf(tot2 * (1.0f / CC_) + 1e-5f);
  const int c = tid * 4;
  half_t* orow = out + (size_t)row * CC_;
  orow[c + 0] = (half_t)(d0 * rs * g[c + 0] + be[c + 0]);
  orow[c + 1] = (half_t)(d1 * rs * g[c + 1] + be[c + 1]);
  orow[c + 2] = (half_t)(d2 * rs * g[c + 2] + be[c + 2]);
  orow[c + 3] = (half_t)(d3 * rs * g[c + 3] + be[c + 3]);
}

// ---------------------------------------------------------------------------
// Tiled WMMA GEMM: C[M,N] = A[M,K](f16) * Wt[N,K](f16, pre-transposed),
// 128x128x32 tiles, double-buffered LDS fed by async global->LDS B128 copies.
// 256 threads = 8 waves (4x2), wave tile 32x64 = 2x4 accumulators.
// EPI 0: split QKV -> q/k/v [B,H,T,D] f16
// EPI 1: + bias + residual -> f32 out
// EPI 2: + bias, exact GELU -> f16 out
// ---------------------------------------------------------------------------
template <int EPI>
__global__ __launch_bounds__(256) void gemm_wmma(
    const half_t* __restrict__ A, const half_t* __restrict__ Bt,
    const float* __restrict__ bias, const float* __restrict__ res,
    float* __restrict__ outF, half_t* __restrict__ outH,
    half_t* __restrict__ q16, half_t* __restrict__ k16, half_t* __restrict__ v16p,
    int N, int K) {
  __shared__ half_t As[2][128 * 40];   // [m][k], stride 40 (bank-spread)
  __shared__ half_t Bs[2][128 * 40];   // [n][k]

  const int tid = threadIdx.x;
  const int wave = tid >> 5;
  const int lane = tid & 31;
  const int bm = blockIdx.y * 128;
  const int bn = blockIdx.x * 128;
  const int wm = (wave >> 1) * 32;
  const int wn = (wave & 1) * 64;
  const int base8 = tid * 8;

  const v8f zero = {};
  v8f acc[2][4];
#pragma unroll
  for (int i = 0; i < 2; ++i)
#pragma unroll
    for (int j = 0; j < 4; ++j) acc[i][j] = zero;

  auto issue = [&](int k0, int buf) {
#pragma unroll
    for (int p = 0; p < 2; ++p) {
      const int idx = p * 2048 + base8;
      const int rr = idx >> 5, cc = idx & 31;      // same decomposition for A & B
      async_copy_b128(A  + (size_t)(bm + rr) * K + k0 + cc, &As[buf][rr * 40 + cc]);
      async_copy_b128(Bt + (size_t)(bn + rr) * K + k0 + cc, &Bs[buf][rr * 40 + cc]);
    }
  };

  issue(0, 0);
  for (int k0 = 0; k0 < K; k0 += 32) {
    const int cur = (k0 >> 5) & 1;
    async_wait0();          // this wave's async writes into buf `cur` landed
    __syncthreads();        // everyone's landed; prior reads of buf cur^1 done
    if (k0 + 32 < K) issue(k0 + 32, cur ^ 1);

    v16h af[2], bf[4];
#pragma unroll
    for (int i = 0; i < 2; ++i) af[i] = frag_a_lds(&As[cur][(wm + 16 * i) * 40], 40);
#pragma unroll
    for (int j = 0; j < 4; ++j) bf[j] = frag_b_lds(&Bs[cur][(wn + 16 * j) * 40], 40);
#pragma unroll
    for (int i = 0; i < 2; ++i)
#pragma unroll
      for (int j = 0; j < 4; ++j)
        acc[i][j] = __builtin_amdgcn_wmma_f32_16x16x32_f16(
            false, af[i], false, bf[j], (short)0, acc[i][j], false, false);
  }

  // Epilogue: C-layout lane->element mapping (ISA 7.12.2)
  const int rowSel = (lane >> 4) * 8;
  const int colSel = lane & 15;
#pragma unroll
  for (int i = 0; i < 2; ++i) {
#pragma unroll
    for (int j = 0; j < 4; ++j) {
#pragma unroll
      for (int r = 0; r < 8; ++r) {
        const int gm = bm + wm + 16 * i + rowSel + r;
        const int gn = bn + wn + 16 * j + colSel;
        float val = acc[i][j][r] + bias[gn];
        if constexpr (EPI == 0) {
          const int which = gn >> 10;          // 0=q 1=k 2=v
          const int hh = (gn >> 6) & 15;
          const int dd = gn & 63;
          const int b = gm >> 11;
          const int t = gm & 2047;
          half_t* dst = (which == 0) ? q16 : ((which == 1) ? k16 : v16p);
          dst[((size_t)(b * HH_ + hh) * TT_ + t) * DD_ + dd] = (half_t)val;
        } else if constexpr (EPI == 1) {
          val += res[(size_t)gm * N + gn];
          outF[(size_t)gm * N + gn] = val;
        } else {
          const float gl = 0.5f * val * (1.0f + erff(val * 0.70710678118654752f));
          outH[(size_t)gm * N + gn] = (half_t)gl;
        }
      }
    }
  }
}

// ---------------------------------------------------------------------------
// Flash attention, causal, one block = (b, h, 128 query rows), wave = 16 rows
// K/V blocks double-buffered via async global->LDS copies.
// ---------------------------------------------------------------------------
__global__ __launch_bounds__(256) void attn_kernel(const half_t* __restrict__ Q,
                                                   const half_t* __restrict__ Kt,
                                                   const half_t* __restrict__ V,
                                                   half_t* __restrict__ Y) {
  __shared__ half_t Qs[128 * 72];
  __shared__ half_t Ks[2][32 * 72];
  __shared__ half_t Vs[2][32 * 72];
  __shared__ half_t Ps[8][16 * 40];   // per-wave P scratch (C-layout -> A-frag)

  const int tid = threadIdx.x;
  const int wave = tid >> 5;
  const int lane = tid & 31;
  const int qTile = blockIdx.x * 128;
  const int hh = blockIdx.y;
  const int b = blockIdx.z;
  const size_t headBase = (size_t)(b * HH_ + hh) * TT_ * DD_;

  // stage Q tile (128 x 64 f16) via async DMA
  {
    const int r = tid >> 3, c = (tid & 7) * 8;
#pragma unroll
    for (int p = 0; p < 4; ++p)
      async_copy_b128(Q + headBase + (size_t)(qTile + p * 32 + r) * DD_ + c,
                      &Qs[(p * 32 + r) * 72 + c]);
  }

  const int kvR = tid >> 3, kvC = (tid & 7) * 8;
  auto issueKV = [&](int kb, int buf) {
    async_copy_b128(Kt + headBase + (size_t)(kb + kvR) * DD_ + kvC,
                    &Ks[buf][kvR * 72 + kvC]);
    async_copy_b128(V + headBase + (size_t)(kb + kvR) * DD_ + kvC,
                    &Vs[buf][kvR * 72 + kvC]);
  };
  issueKV(0, 0);        // prefetch first K/V block (behind the Q copies)

  async_wait0();        // Q + first K/V landed (this wave's)
  __syncthreads();

  const int q0 = qTile + wave * 16;
  v16h qf[2];
  qf[0] = frag_a_lds(&Qs[wave * 16 * 72], 72);       // d = 0..31
  qf[1] = frag_a_lds(&Qs[wave * 16 * 72 + 32], 72);  // d = 32..63

  const v8f zero = {};
  v8f o[4];
#pragma unroll
  for (int nt = 0; nt < 4; ++nt) o[nt] = zero;
  float mrow[8], lrow[8];
#pragma unroll
  for (int r = 0; r < 8; ++r) { mrow[r] = -1e30f; lrow[r] = 0.f; }

  const int rowSel = (lane >> 4) * 8;
  const int colSel = lane & 15;
  const int kEnd = qTile + 128;

  for (int kb = 0; kb < kEnd; kb += 32) {
    const int cur = (kb >> 5) & 1;
    if (kb + 32 < kEnd) issueKV(kb + 32, cur ^ 1);
    if (kb > q0 + 15) {          // wave-uniform: block entirely in the future
      async_wait0();
      __syncthreads();
      continue;
    }

    // S = Q @ K^T   (contraction over d=64 -> two WMMAs per 16-key tile)
    v8f s[2];
#pragma unroll
    for (int jn = 0; jn < 2; ++jn) {
      const v16h b0 = frag_b_lds(&Ks[cur][jn * 16 * 72], 72);
      const v16h b1 = frag_b_lds(&Ks[cur][jn * 16 * 72 + 32], 72);
      s[jn] = __builtin_amdgcn_wmma_f32_16x16x32_f16(false, qf[0], false, b0,
                                                     (short)0, zero, false, false);
      s[jn] = __builtin_amdgcn_wmma_f32_16x16x32_f16(false, qf[1], false, b1,
                                                     (short)0, s[jn], false, false);
    }
    // scale + causal mask
    const bool needMask = (kb + 31 > q0);
#pragma unroll
    for (int jn = 0; jn < 2; ++jn)
#pragma unroll
      for (int r8 = 0; r8 < 8; ++r8) {
        float sv = s[jn][r8] * 0.125f;   // 1/sqrt(64)
        if (needMask) {
          const int key = kb + jn * 16 + colSel;
          const int qq  = q0 + rowSel + r8;
          if (key > qq) sv = -1e30f;
        }
        s[jn][r8] = sv;
      }
    // online softmax (row stats across the 16 lanes of each half-wave)
#pragma unroll
    for (int r8 = 0; r8 < 8; ++r8) {
      float mx = fmaxf(s[0][r8], s[1][r8]);
#pragma unroll
      for (int off = 8; off >= 1; off >>= 1) mx = fmaxf(mx, __shfl_xor(mx, off, 32));
      const float mnew = fmaxf(mrow[r8], mx);
      const float alpha = __expf(mrow[r8] - mnew);
      const float p0 = __expf(s[0][r8] - mnew);
      const float p1 = __expf(s[1][r8] - mnew);
      float ps = p0 + p1;
#pragma unroll
      for (int off = 8; off >= 1; off >>= 1) ps += __shfl_xor(ps, off, 32);
      lrow[r8] = lrow[r8] * alpha + ps;
      mrow[r8] = mnew;
      Ps[wave][(rowSel + r8) * 40 + colSel]      = (half_t)p0;
      Ps[wave][(rowSel + r8) * 40 + 16 + colSel] = (half_t)p1;
#pragma unroll
      for (int nt = 0; nt < 4; ++nt) o[nt][r8] *= alpha;
    }
    // O += P @ V  (P 16x32, V 32x64 -> four 16x16 output tiles)
    const v16h pf = frag_a_lds(&Ps[wave][0], 40);
#pragma unroll
    for (int nt = 0; nt < 4; ++nt) {
      const v16h vb = frag_b_strided_lds(&Vs[cur][nt * 16], 72);
      o[nt] = __builtin_amdgcn_wmma_f32_16x16x32_f16(false, pf, false, vb,
                                                     (short)0, o[nt], false, false);
    }
    async_wait0();      // next K/V block (issued above) landed
    __syncthreads();    // all waves done reading buf `cur`, next buf visible
  }

  // normalize and write Y [B,T,C] f16
#pragma unroll
  for (int nt = 0; nt < 4; ++nt)
#pragma unroll
    for (int r8 = 0; r8 < 8; ++r8) {
      const int qq = q0 + rowSel + r8;
      const int dd = nt * 16 + colSel;
      const float ov = o[nt][r8] / lrow[r8];
      Y[(size_t)(b * TT_ + qq) * CC_ + hh * DD_ + dd] = (half_t)ov;
    }
}

// ---------------------------------------------------------------------------
// Host orchestration
// ---------------------------------------------------------------------------
extern "C" void kernel_launch(void* const* d_in, const int* in_sizes, int n_in,
                              void* d_out, int out_size, void* d_ws, size_t ws_size,
                              hipStream_t stream) {
  (void)in_sizes; (void)n_in; (void)out_size; (void)ws_size;
  const float* x     = (const float*)d_in[0];
  const float* W_qkv = (const float*)d_in[1];
  const float* b_qkv = (const float*)d_in[2];
  const float* W_ao  = (const float*)d_in[3];
  const float* b_ao  = (const float*)d_in[4];
  const float* W_fc  = (const float*)d_in[5];
  const float* b_fc  = (const float*)d_in[6];
  const float* W_mp  = (const float*)d_in[7];
  const float* b_mp  = (const float*)d_in[8];
  const float* g1    = (const float*)d_in[9];
  const float* be1   = (const float*)d_in[10];
  const float* g2    = (const float*)d_in[11];
  const float* be2   = (const float*)d_in[12];

  char* ws = (char*)d_ws;
  size_t off = 0;
  auto alloc = [&](size_t bytes) -> char* {
    char* p = ws + off;
    off += (bytes + 255) & ~(size_t)255;
    return p;
  };

  const size_t szQKVW = (size_t)CC_ * 3 * CC_;
  const size_t szSqW  = (size_t)CC_ * CC_;
  const size_t szFcW  = (size_t)CC_ * 4 * CC_;
  const size_t szAct  = (size_t)MM_ * CC_;       // 4096 x 1024
  const size_t szMid  = (size_t)MM_ * 4 * CC_;   // 4096 x 4096

  // transposed f16 weights: Wt[N,K]
  half_t* Wqkv16 = (half_t*)alloc(szQKVW * 2);
  half_t* Wao16  = (half_t*)alloc(szSqW * 2);
  half_t* Wfc16  = (half_t*)alloc(szFcW * 2);
  half_t* Wmp16  = (half_t*)alloc(szFcW * 2);
  half_t* h16    = (half_t*)alloc(szAct * 2);
  half_t* q16    = (half_t*)alloc(szAct * 2);
  half_t* k16    = (half_t*)alloc(szAct * 2);
  half_t* v16b   = (half_t*)alloc(szAct * 2);
  half_t* y16    = (half_t*)alloc(szAct * 2);
  float*  x1     = (float*)alloc(szAct * 4);
  half_t* h2_16  = (half_t*)alloc(szAct * 2);
  half_t* m16    = (half_t*)alloc(szMid * 2);

  const dim3 blk(256);
  const dim3 tblk(32, 8);

  transpose_to_f16<<<dim3(3 * CC_ / 32, CC_ / 32), tblk, 0, stream>>>(W_qkv, Wqkv16, CC_, 3 * CC_);
  transpose_to_f16<<<dim3(CC_ / 32, CC_ / 32), tblk, 0, stream>>>(W_ao, Wao16, CC_, CC_);
  transpose_to_f16<<<dim3(4 * CC_ / 32, CC_ / 32), tblk, 0, stream>>>(W_fc, Wfc16, CC_, 4 * CC_);
  transpose_to_f16<<<dim3(CC_ / 32, 4 * CC_ / 32), tblk, 0, stream>>>(W_mp, Wmp16, 4 * CC_, CC_);

  // LN1
  ln_kernel<<<dim3(MM_), blk, 0, stream>>>(x, g1, be1, h16);
  // QKV GEMM: [4096,1024] x [1024,3072] -> q/k/v [B,H,T,D] f16
  gemm_wmma<0><<<dim3(3 * CC_ / 128, MM_ / 128), blk, 0, stream>>>(
      h16, Wqkv16, b_qkv, nullptr, nullptr, nullptr, q16, k16, v16b, 3 * CC_, CC_);
  // causal flash attention -> y16 [B,T,C]
  attn_kernel<<<dim3(TT_ / 128, HH_, BB_), blk, 0, stream>>>(q16, k16, v16b, y16);
  // output projection + residual -> x1 (f32)
  gemm_wmma<1><<<dim3(CC_ / 128, MM_ / 128), blk, 0, stream>>>(
      y16, Wao16, b_ao, x, x1, nullptr, nullptr, nullptr, nullptr, CC_, CC_);
  // LN2
  ln_kernel<<<dim3(MM_), blk, 0, stream>>>(x1, g2, be2, h2_16);
  // FC + GELU -> m16 (f16)
  gemm_wmma<2><<<dim3(4 * CC_ / 128, MM_ / 128), blk, 0, stream>>>(
      h2_16, Wfc16, b_fc, nullptr, nullptr, m16, nullptr, nullptr, nullptr, 4 * CC_, CC_);
  // down-proj + residual -> out (f32)
  gemm_wmma<1><<<dim3(CC_ / 128, MM_ / 128), blk, 0, stream>>>(
      m16, Wmp16, b_mp, x1, (float*)d_out, nullptr, nullptr, nullptr, nullptr, CC_, 4 * CC_);
}
